// MultiHeadAttentionCausal_25091198943732
// MI455X (gfx1250) — matrix operand order repaired and checked by reading, not compiled, not run on hardware
//
#include <hip/hip_runtime.h>
#include <hip/hip_bf16.h>

typedef __bf16 bf16;
typedef __attribute__((ext_vector_type(16))) __bf16 v16bf;
typedef __attribute__((ext_vector_type(8)))  __bf16 bf16x8;
typedef __attribute__((ext_vector_type(8)))  float   v8f;

#define B_  4
#define S_  2048
#define D_  768
#define H_  12
#define HD_ 64

__device__ __forceinline__ v8f zero8() {
  v8f z = {0.f, 0.f, 0.f, 0.f, 0.f, 0.f, 0.f, 0.f};
  return z;
}

__device__ __forceinline__ v8f wmma_bf16(v16bf a, v16bf b, v8f c) {
  return __builtin_amdgcn_wmma_f32_16x16x32_bf16(
      /*neg_a=*/false, a, /*neg_b=*/false, b,
      /*c_mod=*/(short)0, c, /*reuse_a=*/false, /*reuse_b=*/false);
}

// A operand: 16x32 bf16, row-major source, leading dim `ld` (elements).
// lane l (m = l&15, half = l>>4): slots 0..7 -> k = 8*half+0..7,
// slots 8..15 -> k = 16+8*half+0..7 (two 16B contiguous runs).
__device__ __forceinline__ v16bf load_a(const bf16* base, int ld) {
  const int lane = threadIdx.x & 31;
  const int m    = lane & 15;
  const int half = lane >> 4;
  const bf16* p  = base + (size_t)m * ld + 8 * half;
  bf16x8 lo = *(const bf16x8*)(p);
  bf16x8 hi = *(const bf16x8*)(p + 16);
  v16bf f;
#pragma unroll
  for (int i = 0; i < 8; ++i) { f[i] = lo[i]; f[i + 8] = hi[i]; }
  return f;
}

// B operand: 32x16 bf16 (K x N). Source is "B^T row-major": row n holds
// contiguous K, leading dim `ld`. lane l (n = l&15, half = l>>4):
// slots 0..7 -> k = 8*half+0..7, slots 8..15 -> k = 16+8*half+0..7 (two 16B runs).
__device__ __forceinline__ v16bf load_b(const bf16* base, int ld) {
  const int lane = threadIdx.x & 31;
  const int n    = lane & 15;
  const int half = lane >> 4;
  const bf16* p  = base + (size_t)n * ld + 8 * half;
  bf16x8 lo = *(const bf16x8*)(p);
  bf16x8 hi = *(const bf16x8*)(p + 16);
  v16bf f;
#pragma unroll
  for (int i = 0; i < 8; ++i) { f[i] = lo[i]; f[i + 8] = hi[i]; }
  return f;
}

// ---------------- conversion kernels ----------------

__global__ void cvt_f32_bf16(const float* __restrict__ src, bf16* __restrict__ dst, int n) {
  int i = blockIdx.x * blockDim.x + threadIdx.x;
  int stride = gridDim.x * blockDim.x;
  for (; i < n; i += stride) dst[i] = (bf16)src[i];
}

// src: [H, D, HD] f32 -> dst: [H, HD, D] bf16 (transposed per head)
__global__ void cvt_transpose_w(const float* __restrict__ src, bf16* __restrict__ dst) {
  const int total = H_ * HD_ * D_;
  int i = blockIdx.x * blockDim.x + threadIdx.x;
  int stride = gridDim.x * blockDim.x;
  for (; i < total; i += stride) {
    int h   = i / (HD_ * D_);
    int rem = i % (HD_ * D_);
    int e   = rem / D_;
    int d   = rem % D_;
    dst[i] = (bf16)src[(size_t)h * D_ * HD_ + (size_t)d * HD_ + e];
  }
}

// ---------------- QKV projection ----------------
// One wave: 32 x-rows x 64 cols of ONE of {q,k,v} for one head.
// wt3 = packed transposed weights [3][H][HD][D]; q,k row-major [B*H,S,HD],
// v transposed [B*H,HD,S].
__global__ void qkv_kernel(const bf16* __restrict__ xb,
                           const bf16* __restrict__ wt3,
                           bf16* __restrict__ qo,
                           bf16* __restrict__ ko,
                           bf16* __restrict__ vto) {
  const int wave = blockIdx.x * (blockDim.x >> 5) + (threadIdx.x >> 5);
  const int RT = S_ / 32;                       // 64 row tiles
  int t0 = wave;
  const int rt = t0 % RT; t0 /= RT;
  const int bh = t0 % (B_ * H_); t0 /= (B_ * H_);
  const int which = t0;                         // 0=q 1=k 2=v
  const int b = bh / H_;
  const int lane = threadIdx.x & 31;
  const int n = lane & 15, g = lane >> 4;

  const bf16* xrow = xb + ((size_t)b * S_ + (size_t)rt * 32) * D_;
  const bf16* wh   = wt3 + ((size_t)which * H_ + (bh % H_)) * HD_ * D_;

  v8f acc0[4], acc1[4];
#pragma unroll
  for (int t = 0; t < 4; ++t) { acc0[t] = zero8(); acc1[t] = zero8(); }

  for (int d0 = 0; d0 < D_; d0 += 32) {
    v16bf ax0 = load_a(xrow + d0, D_);
    v16bf ax1 = load_a(xrow + (size_t)16 * D_ + d0, D_);
#pragma unroll
    for (int t = 0; t < 4; ++t) {
      v16bf bw = load_b(wh + (size_t)(t * 16) * D_ + d0, D_);
      acc0[t] = wmma_bf16(ax0, bw, acc0[t]);
      acc1[t] = wmma_bf16(ax1, bw, acc1[t]);
    }
  }

  const int s0r = rt * 32;
  if (which == 2) {
    bf16* vh = vto + (size_t)bh * HD_ * S_;
#pragma unroll
    for (int t = 0; t < 4; ++t) {
#pragma unroll
      for (int r = 0; r < 8; ++r) {
        int e = t * 16 + n;
        int m = r + 8 * g;
        vh[(size_t)e * S_ + s0r + m]      = (bf16)acc0[t][r];
        vh[(size_t)e * S_ + s0r + 16 + m] = (bf16)acc1[t][r];
      }
    }
  } else {
    bf16* orow = (which == 0 ? qo : ko) + ((size_t)bh * S_ + s0r) * HD_;
#pragma unroll
    for (int t = 0; t < 4; ++t) {
#pragma unroll
      for (int r = 0; r < 8; ++r) {
        int m = r + 8 * g;
        orow[(size_t)m * HD_ + t * 16 + n]        = (bf16)acc0[t][r];
        orow[(size_t)(16 + m) * HD_ + t * 16 + n] = (bf16)acc1[t][r];
      }
    }
  }
}

// ---------------- flash attention (causal, online softmax) ----------------
// One wave: 16 query rows for one (b,h); loop over 32-key tiles.
__global__ void attn_kernel(const bf16* __restrict__ q,
                            const bf16* __restrict__ k,
                            const bf16* __restrict__ vt,
                            bf16* __restrict__ ctx) {
  __shared__ __align__(16) bf16 plds[4][16 * 32];
  const int wslot = threadIdx.x >> 5;
  const int wave  = blockIdx.x * 4 + wslot;
  const int QT = S_ / 16;
  const int qt = wave % QT;
  const int bh = wave / QT;
  const int b  = bh / H_;
  const int h  = bh % H_;
  const int lane = threadIdx.x & 31;
  const int n = lane & 15, g = lane >> 4;
  const int q_base = qt * 16;

  const bf16* qrow = q + ((size_t)bh * S_ + q_base) * HD_;
  v16bf aq0 = load_a(qrow, HD_);
  v16bf aq1 = load_a(qrow + 32, HD_);

  v8f acc[4];
  float mi[8], li[8];
#pragma unroll
  for (int t = 0; t < 4; ++t) acc[t] = zero8();
#pragma unroll
  for (int r = 0; r < 8; ++r) { mi[r] = -3.0e38f; li[r] = 0.f; }

  const bf16* kbase = k + (size_t)bh * S_ * HD_;
  const bf16* vbase = vt + (size_t)bh * HD_ * S_;
  bf16* pl = plds[wslot];

  for (int kt = 0; kt <= q_base + 15; kt += 32) {
    // K fragments for both 16-key subtiles (one big load clause)
    v16bf b00 = load_b(kbase + (size_t)kt * HD_, HD_);
    v16bf b01 = load_b(kbase + (size_t)kt * HD_ + 32, HD_);
    v16bf b10 = load_b(kbase + (size_t)(kt + 16) * HD_, HD_);
    v16bf b11 = load_b(kbase + (size_t)(kt + 16) * HD_ + 32, HD_);

    v8f s0 = zero8(), s1 = zero8();
    s0 = wmma_bf16(aq0, b00, s0);
    s0 = wmma_bf16(aq1, b01, s0);
    s1 = wmma_bf16(aq0, b10, s1);
    s1 = wmma_bf16(aq1, b11, s1);

    // Issue V-fragment loads early: independent of softmax, so their latency
    // overlaps the shuffle/exp VALU work below.
    v16bf bv0 = load_b(vbase + (size_t)(0 * 16) * S_ + kt, S_);
    v16bf bv1 = load_b(vbase + (size_t)(1 * 16) * S_ + kt, S_);
    v16bf bv2 = load_b(vbase + (size_t)(2 * 16) * S_ + kt, S_);
    v16bf bv3 = load_b(vbase + (size_t)(3 * 16) * S_ + kt, S_);

    // scale + causal mask + online softmax. Row m = q_base + r + 8*g; its 16
    // elements live across the 16 lanes of this lane-group -> shfl_xor reduce.
    float p0[8], p1[8];
#pragma unroll
    for (int r = 0; r < 8; ++r) {
      int m = q_base + r + 8 * g;
      float x0 = s0[r] * 0.125f;            // 1/sqrt(64)
      float x1 = s1[r] * 0.125f;
      if (kt + n      > m) x0 = -3.0e38f;
      if (kt + 16 + n > m) x1 = -3.0e38f;
      float rm = fmaxf(x0, x1);
#pragma unroll
      for (int msk = 1; msk < 16; msk <<= 1) rm = fmaxf(rm, __shfl_xor(rm, msk, 32));
      float mnew = fmaxf(mi[r], rm);
      float sc = __expf(mi[r] - mnew);
      float e0 = __expf(x0 - mnew);
      float e1 = __expf(x1 - mnew);
      float rs = e0 + e1;
#pragma unroll
      for (int msk = 1; msk < 16; msk <<= 1) rs += __shfl_xor(rs, msk, 32);
      li[r] = li[r] * sc + rs;
      mi[r] = mnew;
#pragma unroll
      for (int t = 0; t < 4; ++t) acc[t][r] *= sc;
      p0[r] = e0;
      p1[r] = e1;
    }

    // D-layout f32 -> row-major bf16 in LDS (same wave; compiler inserts
    // s_wait_dscnt for the RAW dependence, no barrier needed)
#pragma unroll
    for (int r = 0; r < 8; ++r) {
      int m = r + 8 * g;
      pl[m * 32 + n]      = (bf16)p0[r];
      pl[m * 32 + 16 + n] = (bf16)p1[r];
    }
    v16bf pA = load_a(pl, 32);

    acc[0] = wmma_bf16(pA, bv0, acc[0]);
    acc[1] = wmma_bf16(pA, bv1, acc[1]);
    acc[2] = wmma_bf16(pA, bv2, acc[2]);
    acc[3] = wmma_bf16(pA, bv3, acc[3]);
  }

  // normalize and store ctx as bf16 [B, S, H*HD]
  bf16* crow = ctx + ((size_t)b * S_ + q_base) * D_ + (size_t)h * HD_;
#pragma unroll
  for (int t = 0; t < 4; ++t) {
#pragma unroll
    for (int r = 0; r < 8; ++r) {
      int m = r + 8 * g;
      float v = acc[t][r] / li[r];
      crow[(size_t)m * D_ + t * 16 + n] = (bf16)v;
    }
  }
}

// ---------------- output projection: out = ctx @ Wo^T + bo ----------------
// One wave: 32 rows x 64 dout cols.
__global__ void oproj_kernel(const bf16* __restrict__ ctx,
                             const bf16* __restrict__ wob,
                             const float* __restrict__ bo,
                             float* __restrict__ out) {
  const int wave = blockIdx.x * (blockDim.x >> 5) + (threadIdx.x >> 5);
  const int CT = D_ / 64;                     // 12 col tiles
  const int ct = wave % CT;
  const int rt = wave / CT;                   // 0..255 row tiles of 32
  const int lane = threadIdx.x & 31;
  const int n = lane & 15, g = lane >> 4;

  const bf16* arow  = ctx + (size_t)rt * 32 * D_;
  const bf16* wbase = wob + (size_t)ct * 64 * D_;  // row dout: contiguous din

  v8f acc0[4], acc1[4];
#pragma unroll
  for (int t = 0; t < 4; ++t) { acc0[t] = zero8(); acc1[t] = zero8(); }

  for (int d0 = 0; d0 < D_; d0 += 32) {
    v16bf a0 = load_a(arow + d0, D_);
    v16bf a1 = load_a(arow + (size_t)16 * D_ + d0, D_);
#pragma unroll
    for (int t = 0; t < 4; ++t) {
      v16bf bw = load_b(wbase + (size_t)(t * 16) * D_ + d0, D_);
      acc0[t] = wmma_bf16(a0, bw, acc0[t]);
      acc1[t] = wmma_bf16(a1, bw, acc1[t]);
    }
  }

  float* orow = out + (size_t)rt * 32 * D_ + (size_t)ct * 64;
#pragma unroll
  for (int t = 0; t < 4; ++t) {
    float bias = bo[ct * 64 + t * 16 + n];
#pragma unroll
    for (int r = 0; r < 8; ++r) {
      int m = r + 8 * g;
      orow[(size_t)m * D_ + t * 16 + n]        = acc0[t][r] + bias;
      orow[(size_t)(16 + m) * D_ + t * 16 + n] = acc1[t][r] + bias;
    }
  }
}

// ---------------- launch ----------------

extern "C" void kernel_launch(void* const* d_in, const int* in_sizes, int n_in,
                              void* d_out, int out_size, void* d_ws, size_t ws_size,
                              hipStream_t stream) {
  (void)in_sizes; (void)n_in; (void)out_size; (void)ws_size;
  const float* x  = (const float*)d_in[0];
  const float* Wq = (const float*)d_in[1];
  const float* Wk = (const float*)d_in[2];
  const float* Wv = (const float*)d_in[3];
  const float* Wo = (const float*)d_in[4];
  const float* bo = (const float*)d_in[5];
  float* out = (float*)d_out;

  char* ws = (char*)d_ws;
  const size_t SZ_X   = (size_t)B_ * S_ * D_ * 2;        // 12,582,912
  const size_t SZ_W   = (size_t)H_ * HD_ * D_ * 2;       //  1,179,648
  const size_t SZ_WO  = (size_t)D_ * D_ * 2;             //  1,179,648
  const size_t SZ_QKV = (size_t)B_ * H_ * S_ * HD_ * 2;  // 12,582,912

  bf16* xb   = (bf16*)(ws);
  bf16* wt3  = (bf16*)(ws + SZ_X);                        // [3][H][HD][D]
  bf16* wob  = (bf16*)(ws + SZ_X + 3 * SZ_W);
  bf16* qb   = (bf16*)(ws + SZ_X + 3 * SZ_W + SZ_WO);
  bf16* kb   = (bf16*)(ws + SZ_X + 3 * SZ_W + SZ_WO + SZ_QKV);
  bf16* vtb  = (bf16*)(ws + SZ_X + 3 * SZ_W + SZ_WO + 2 * SZ_QKV);
  bf16* ctxb = (bf16*)(ws + SZ_X + 3 * SZ_W + SZ_WO + 3 * SZ_QKV);

  cvt_f32_bf16<<<2048, 256, 0, stream>>>(x, xb, B_ * S_ * D_);
  cvt_f32_bf16<<<512, 256, 0, stream>>>(Wo, wob, D_ * D_);
  cvt_transpose_w<<<512, 256, 0, stream>>>(Wq, wt3);
  cvt_transpose_w<<<512, 256, 0, stream>>>(Wk, wt3 + (size_t)H_ * HD_ * D_);
  cvt_transpose_w<<<512, 256, 0, stream>>>(Wv, wt3 + (size_t)2 * H_ * HD_ * D_);

  const int qkv_waves = 3 * B_ * H_ * (S_ / 32);         // 9216
  qkv_kernel<<<qkv_waves / 4, 128, 0, stream>>>(xb, wt3, qb, kb, vtb);

  const int attn_waves = B_ * H_ * (S_ / 16);            // 6144
  attn_kernel<<<attn_waves / 4, 128, 0, stream>>>(qb, kb, vtb, ctxb);

  const int op_waves = (B_ * S_ / 32) * (D_ / 64);       // 3072
  oproj_kernel<<<op_waves / 4, 128, 0, stream>>>(ctxb, wob, bo, out);
}
